// ManualRNNModel_49134425866789
// MI455X (gfx1250) — compile-verified, matrix operand order checked by reading, not run on hardware
//
#include <hip/hip_runtime.h>
#include <hip/hip_bf16.h>
#include <math.h>

// Problem constants (from reference): B=256, T=512, I=64, H=512
#define B_   256
#define T_   512
#define I_   64
#define H_   512
#define LDH  520   // h-tile LDS row stride (elements): 1040 B = 260 dwords == 4 mod 64 banks
#define KREG 8     // k-steps (K=0..255) of W_hh^T held in VGPRs
#define KLDS 8     // k-steps (K=256..511) of W_hh^T held in LDS
#define LDB  264   // W_hh LDS row stride (elements): 528 B = 132 dwords == 4 mod 64 banks

typedef __bf16 bf16x16 __attribute__((ext_vector_type(16)));
typedef float  f32x8   __attribute__((ext_vector_type(8)));

union FragU { bf16x16 v; uint4 q[2]; };

// WMMA operand fragment (A layout; equals B layout when fed W^T rows).
// Lane (lr, hi): elements K = k0+hi*8+[0..7] and K = k0+16+hi*8+[0..7]
// -> two contiguous 16-byte chunks, 32 bytes apart.
__device__ __forceinline__ bf16x16 load_frag_b16(const unsigned short* base,
                                                 int row, int ld, int k0, int hi) {
    FragU f;
    const unsigned short* p = base + (size_t)row * ld + k0 + hi * 8;
    f.q[0] = *(const uint4*)(p);
    f.q[1] = *(const uint4*)(p + 16);
    return f.v;
}

// Same fragment pattern from f32 memory (x), converting to bf16 on the fly.
__device__ __forceinline__ bf16x16 load_frag_f32(const float* __restrict__ rowptr, int k0, int hi) {
    const float* p = rowptr + k0 + hi * 8;
    float4 c0 = *(const float4*)(p);
    float4 c1 = *(const float4*)(p + 4);
    float4 c2 = *(const float4*)(p + 16);
    float4 c3 = *(const float4*)(p + 20);
    bf16x16 v;
    v[0]  = (__bf16)c0.x; v[1]  = (__bf16)c0.y; v[2]  = (__bf16)c0.z; v[3]  = (__bf16)c0.w;
    v[4]  = (__bf16)c1.x; v[5]  = (__bf16)c1.y; v[6]  = (__bf16)c1.z; v[7]  = (__bf16)c1.w;
    v[8]  = (__bf16)c2.x; v[9]  = (__bf16)c2.y; v[10] = (__bf16)c2.z; v[11] = (__bf16)c2.w;
    v[12] = (__bf16)c3.x; v[13] = (__bf16)c3.y; v[14] = (__bf16)c3.z; v[15] = (__bf16)c3.w;
    return v;
}

__device__ __forceinline__ float fast_tanh(float x) {
#if __has_builtin(__builtin_amdgcn_tanhf)
    return __builtin_amdgcn_tanhf(x);          // v_tanh_f32 (CDNA5 trans op)
#elif __has_builtin(__builtin_amdgcn_tanh_f32)
    return __builtin_amdgcn_tanh_f32(x);
#else
    return tanhf(x);
#endif
}

// Prep: dst[c*rows + r] = bf16(src[r*cols + c])   (transpose + convert)
__global__ void transpose_cvt_bf16(const float* __restrict__ src, unsigned short* __restrict__ dst,
                                   int rows, int cols) {
    int idx = blockIdx.x * blockDim.x + threadIdx.x;
    if (idx >= rows * cols) return;
    int c = idx / rows;
    int r = idx % rows;
    float v = src[(size_t)r * cols + c];
    dst[idx] = __builtin_bit_cast(unsigned short, (__bf16)v);
}

// Persistent RNN scan: block = 8 waves = 256 threads, owns 16 batch rows for all T.
// Wave w owns column tiles {w, w+8, w+16, w+24}. W_hh^T is fully on-chip resident:
// K 0..255 in VGPRs (256 regs/wave), K 256..511 in LDS (264 KB).
// A (h) and LDS-B fragments are software double-buffered to hide DS latency under WMMA.
__attribute__((amdgpu_waves_per_eu(2)))
__global__ __launch_bounds__(256) void rnn_scan_kernel(
    const float* __restrict__ x,              // (B, T, I) f32
    const unsigned short* __restrict__ wxhT,  // (H, I) bf16  == W_xh^T
    const unsigned short* __restrict__ whhT,  // (H, H) bf16  == W_hh^T
    const float* __restrict__ bias,           // (H) f32
    const float* __restrict__ fc_w,           // (H) f32
    const float* __restrict__ fc_b,           // (1) f32
    float* __restrict__ out)                  // (B) f32
{
    __shared__ unsigned short hbuf[2][16 * LDH];   // 33,280 B : double-buffered h tile (bf16)
    __shared__ unsigned short blds[H_ * LDB];      // 270,336 B: W_hh^T rows, K-half 256..511

    const int tid  = threadIdx.x;
    const int wave = tid >> 5;     // 0..7
    const int lane = tid & 31;
    const int lr   = lane & 15;
    const int hi   = lane >> 4;
    const int rb   = blockIdx.x * 16;

    // zero h0 (both buffers incl. padding)
    {
        unsigned short* hb = &hbuf[0][0];
        for (int i = tid; i < 2 * 16 * LDH; i += 256) hb[i] = 0;
    }

    // One-time preload of W_hh^T upper K-half into LDS (padded rows), 16B vector copies.
    {
        const uint4* src = (const uint4*)whhT;     // global row stride = 64 uint4
        for (int idx = tid; idx < H_ * 32; idx += 256) {
            int row = idx >> 5;
            int c   = idx & 31;                    // 32 uint4 per 256-element row-half
            *(uint4*)(&blds[row * LDB + c * 8]) = src[row * 64 + 32 + c];
        }
    }

    int ncol[4];
    #pragma unroll
    for (int j = 0; j < 4; ++j) ncol[j] = (wave + 8 * j) * 16;

    // Register-resident W_hh^T fragments: K 0..255 (4 tiles x 8 k-steps x 8 VGPRs).
    bf16x16 breg[4][KREG];
    #pragma unroll
    for (int j = 0; j < 4; ++j)
        #pragma unroll
        for (int ks = 0; ks < KREG; ++ks)
            breg[j][ks] = load_frag_b16(whhT, ncol[j] + lr, H_, ks * 32, hi);

    // t-invariant W_xh^T fragments (K=64 -> 2 k-steps per tile).
    bf16x16 bx[4][2];
    #pragma unroll
    for (int j = 0; j < 4; ++j)
        #pragma unroll
        for (int ks = 0; ks < 2; ++ks)
            bx[j][ks] = load_frag_b16(wxhT, ncol[j] + lr, I_, ks * 32, hi);

    float bias_[4];
    #pragma unroll
    for (int j = 0; j < 4; ++j) bias_[j] = bias[ncol[j] + lr];

    __syncthreads();

    int cur = 0;
    for (int t = 0; t < T_; ++t) {
        f32x8 acc[4] = {};   // bias folded into the tanh epilogue; SRC2 starts as inline 0

        const unsigned short* hcur = &hbuf[cur][0];

        // Issue long-latency x loads first, then warm the A pipeline.
        const float* xrow = x + ((size_t)(rb + lr) * T_ + t) * I_;
        __builtin_prefetch(xrow + I_, 0, 3);       // next timestep's row -> global_prefetch
        bf16x16 ax0 = load_frag_f32(xrow, 0, hi);
        bf16x16 ax1 = load_frag_f32(xrow, 32, hi);
        bf16x16 a_cur = load_frag_b16(hcur, lr, LDH, 0, hi);

        // ---- input projection: x_t(16x64) @ W_xh tiles
        #pragma unroll
        for (int j = 0; j < 4; ++j)
            acc[j] = __builtin_amdgcn_wmma_f32_16x16x32_bf16(false, ax0, false, bx[j][0], (short)0, acc[j], false, false);
        #pragma unroll
        for (int j = 0; j < 4; ++j)
            acc[j] = __builtin_amdgcn_wmma_f32_16x16x32_bf16(false, ax1, false, bx[j][1], (short)0, acc[j], false, false);

        // ---- recurrent, K 0..255: B from registers; A double-buffered
        #pragma unroll
        for (int ks = 0; ks < KREG; ++ks) {
            bf16x16 a_nxt = load_frag_b16(hcur, lr, LDH, (ks + 1) * 32, hi);  // ks+1 <= 8 valid
            #pragma unroll
            for (int j = 0; j < 4; ++j)
                acc[j] = __builtin_amdgcn_wmma_f32_16x16x32_bf16(false, a_cur, false, breg[j][ks], (short)0, acc[j], false, false);
            a_cur = a_nxt;
        }

        // ---- recurrent, K 256..511: B from LDS; both A and B double-buffered
        bf16x16 bcur[4], bnxt[4];
        #pragma unroll
        for (int j = 0; j < 4; ++j)
            bcur[j] = load_frag_b16(&blds[0], ncol[j] + lr, LDB, 0, hi);
        #pragma unroll
        for (int ks = 0; ks < KLDS; ++ks) {
            bf16x16 a_nxt;
            if (ks < KLDS - 1) {
                a_nxt = load_frag_b16(hcur, lr, LDH, (KREG + ks + 1) * 32, hi);
                #pragma unroll
                for (int j = 0; j < 4; ++j)
                    bnxt[j] = load_frag_b16(&blds[0], ncol[j] + lr, LDB, (ks + 1) * 32, hi);
            }
            #pragma unroll
            for (int j = 0; j < 4; ++j)
                acc[j] = __builtin_amdgcn_wmma_f32_16x16x32_bf16(false, a_cur, false, bcur[j], (short)0, acc[j], false, false);
            if (ks < KLDS - 1) {
                a_cur = a_nxt;
                #pragma unroll
                for (int j = 0; j < 4; ++j) bcur[j] = bnxt[j];
            }
        }

        // ---- h_new = tanh(acc + bias) -> other buffer (row m = hi*8+r, col = ncol[j]+lr)
        unsigned short* hn = &hbuf[cur ^ 1][0];
        #pragma unroll
        for (int r = 0; r < 8; ++r) {
            const int m = hi * 8 + r;
            #pragma unroll
            for (int j = 0; j < 4; ++j) {
                float v = fast_tanh(acc[j][r] + bias_[j]);
                hn[m * LDH + ncol[j] + lr] = __builtin_bit_cast(unsigned short, (__bf16)v);
            }
        }
        __syncthreads();
        cur ^= 1;
    }

    // ---- final FC: out[rb+row] = dot(h_final[row,:], fc_w) + fc_b ; wave handles rows {w, w+8}
    #pragma unroll
    for (int rr = 0; rr < 2; ++rr) {
        const int row = wave + 8 * rr;
        const unsigned short* hr = &hbuf[cur][row * LDH];
        float s = 0.f;
        for (int j = lane; j < H_; j += 32) {
            float hv = (float)__builtin_bit_cast(__bf16, hr[j]);
            s += hv * fc_w[j];
        }
        #pragma unroll
        for (int off = 16; off; off >>= 1) s += __shfl_xor(s, off, 32);
        if (lane == 0) out[rb + row] = s + fc_b[0];
    }
}

extern "C" void kernel_launch(void* const* d_in, const int* in_sizes, int n_in,
                              void* d_out, int out_size, void* d_ws, size_t ws_size,
                              hipStream_t stream) {
    const float* x    = (const float*)d_in[0];   // (256,512,64)
    const float* wxh  = (const float*)d_in[1];   // (64,512)
    const float* whh  = (const float*)d_in[2];   // (512,512)
    const float* bias = (const float*)d_in[3];   // (512)
    const float* fcw  = (const float*)d_in[4];   // (1,512)
    const float* fcb  = (const float*)d_in[5];   // (1)
    float* out = (float*)d_out;                  // (256,1)

    // Workspace: W_xh^T bf16 (512*64), then W_hh^T bf16 (512*512). 576 KB total.
    unsigned short* wxhT = (unsigned short*)d_ws;
    unsigned short* whhT = wxhT + (size_t)H_ * I_;

    transpose_cvt_bf16<<<(I_ * H_ + 255) / 256, 256, 0, stream>>>(wxh, wxhT, I_, H_);
    transpose_cvt_bf16<<<(H_ * H_ + 255) / 256, 256, 0, stream>>>(whh, whhT, H_, H_);
    rnn_scan_kernel<<<B_ / 16, 256, 0, stream>>>(x, wxhT, whhT, bias, fcw, fcb, out);
}
